// UniversalMoEContainer_38534446579730
// MI455X (gfx1250) — matrix operand order repaired and checked by reading, not compiled, not run on hardware
//
#include <hip/hip_runtime.h>

// ---------------------------------------------------------------------------
// Dense-MoE conv3x3 + BN + SiLU + routing-weighted sum, implicit-GEMM on
// CDNA5 WMMA (v_wmma_f32_16x16x32_bf16).
//   B=64, C=32, H=W=128, E=4 experts, top-K=2 routing slots.
//   GEMM per expert:  D[32 x Npix] = Wc[32 x 288] * im2col(x)[288 x Npix]
//   K = 288 = 9 taps * 32 input channels  -> 9 WMMA K-chunks of 32.
//
// Two kernels:
//   1) prep: Wc (fp32) -> bf16, pre-swizzled to a contiguous-K 16-bit A-matrix VGPR
//      layout, written once to d_ws (73,728 B). Runs every launch (stateless).
//   2) main: per block = one (sample, row-pair). Weights are a straight
//      b128 memcpy ws->LDS; x tile converted fp32->bf16 into LDS in the
//      B-fragment-friendly [row][col][ci] order. 8 waves x 32 pixels,
//      4 WMMAs per K-chunk per wave (2 M-tiles x 2 N-tiles).
// ---------------------------------------------------------------------------

#define BDIM 256
constexpr int Bn = 64, Cc = 32, Hh = 128, Ww = 128;
constexpr float BN_EPS = 1e-5f;

// swizzled weight records: [e][mtile][kchunk][lane][16 bf16] = 2304 records
constexpr int WREC = 4 * 2 * 9 * 32;          // 2304
constexpr int WSW_BYTES = WREC * 16 * 2;      // 73,728 B in d_ws

typedef __bf16 v16bf __attribute__((ext_vector_type(16)));
typedef float  v8f   __attribute__((ext_vector_type(8)));

static __device__ __forceinline__ unsigned short f2bf(float f) {
    union { float f; unsigned u; } c; c.f = f;
    unsigned u = c.u;
    unsigned r = u + 0x7FFFu + ((u >> 16) & 1u);   // round-to-nearest-even
    return (unsigned short)(r >> 16);
}

// ---------------------------------------------------------------------------
// Prep: swizzle Wc into A-matrix fragment layout (bf16) in workspace.
// 16-bit A 16x32 layout (contiguous K): lanes 0-15: V0-7 -> K0..15
//                                       lanes16-31: V0-7 -> K16..31
// ---------------------------------------------------------------------------
__global__ __launch_bounds__(BDIM)
void moe_prep_weights(const float* __restrict__ Wc,
                      unsigned short* __restrict__ wsw) {
    int rec = blockIdx.x * BDIM + threadIdx.x;   // [e][mt][kc][lane]
    if (rec >= WREC) return;
    int lane = rec & 31;
    int kc   = (rec >> 5) % 9;
    int mt   = (rec / (9 * 32)) & 1;
    int e    = rec / (2 * 9 * 32);
    int lHi  = lane >> 4;
    int co   = mt * 16 + (lane & 15);
    int ky   = kc / 3, kx = kc % 3;
    unsigned pk[8];
#pragma unroll
    for (int v = 0; v < 8; ++v) {
        int cib = lHi * 16 + v * 2;   // K = ci
        const float* wp =
            Wc + ((((size_t)e * 32 + co) * 32 + cib) * 9) + ky * 3 + kx;
        pk[v] = (unsigned)f2bf(wp[0]) | ((unsigned)f2bf(wp[9]) << 16);
    }
    unsigned* dst = (unsigned*)(wsw + (size_t)rec * 16);
#pragma unroll
    for (int v = 0; v < 8; ++v) dst[v] = pk[v];
}

// ---------------------------------------------------------------------------
// Main kernel: one block per (sample, output-row-pair). 8 wave32 waves;
// wave w: output row h0 + (w>>2), pixels (w&3)*32 .. +31, all 32 out-ch.
// ---------------------------------------------------------------------------
__global__ __launch_bounds__(BDIM)
void moe_conv_bn_silu_kernel(const float* __restrict__ x,
                             const float* __restrict__ route_w,
                             const int*   __restrict__ route_i,
                             const unsigned short* __restrict__ wsw,
                             const float* __restrict__ gamma,
                             const float* __restrict__ beta,
                             const float* __restrict__ mean,
                             const float* __restrict__ var,
                             float* __restrict__ out) {
    // x tile: rows h0-1..h0+2, cols -1..128, 32 ch, bf16 [row][col][ci]
    __shared__ __attribute__((aligned(16))) unsigned short xlds[4][130][32];      // 33,280 B
    // all experts' swizzled weights: [e][mt][kc][lane][16 bf16]
    __shared__ __attribute__((aligned(16))) unsigned short alds[4][2][9][32][16]; // 73,728 B
    __shared__ float sclds[4][32];
    __shared__ float shlds[4][32];

    const int tid = threadIdx.x;
    const int b   = blockIdx.x >> 6;          // sample
    const int h0  = (blockIdx.x & 63) * 2;    // first of 2 output rows

    // ---- BN affine tables ----
    if (tid < 128) {
        int e = tid >> 5, c = tid & 31;
        float s = gamma[e * 32 + c] * rsqrtf(var[e * 32 + c] + BN_EPS);
        sclds[e][c] = s;
        shlds[e][c] = beta[e * 32 + c] - mean[e * 32 + c] * s;
    }

    // ---- weights: straight b128 copy ws -> LDS (pre-swizzled bf16) ----
    {
        const uint4* wsrc = (const uint4*)wsw;
        uint4* wdst = (uint4*)&alds[0][0][0][0][0];
        for (int i = tid; i < WSW_BYTES / 16; i += BDIM)   // 4608 uint4
            wdst[i] = wsrc[i];
    }

    // ---- stage x tile (fp32 -> bf16, [row][col][ci]) ----
    for (int i = tid; i < 4 * 130 * 16; i += BDIM) {       // 8320 dword stores
        int col = i % 130;
        int r   = (i / 130) & 3;
        int p   = i / 520;                                 // channel pair
        int gh  = h0 + r - 1;
        int gw  = col - 1;
        unsigned v = 0;
        if (gh >= 0 && gh < Hh && gw >= 0 && gw < Ww) {
            const float* bp = x + (((size_t)b * Cc + 2 * p) * Hh + gh) * Ww + gw;
            v = (unsigned)f2bf(bp[0]) | ((unsigned)f2bf(bp[(size_t)Hh * Ww]) << 16);
        }
        *(unsigned*)&xlds[r][col][2 * p] = v;
    }

    // ---- per-sample combined routing weight per expert ----
    float wroute[4] = {0.f, 0.f, 0.f, 0.f};
    {
        float w0 = route_w[b * 2 + 0]; int i0 = route_i[b * 2 + 0] & 3;
        float w1 = route_w[b * 2 + 1]; int i1 = route_i[b * 2 + 1] & 3;
        wroute[i0] += w0;
        wroute[i1] += w1;
    }

    const int lane = tid & 31;
    const int wv   = tid >> 5;
    const int n    = lane & 15;
    const int lHi  = lane >> 4;
    const int rsel = wv >> 2;            // which of the 2 output rows
    const int cb   = (wv & 3) * 32;      // wave's first pixel column
    const int h    = h0 + rsel;

    v8f o00 = {}, o01 = {}, o10 = {}, o11 = {};   // [mtile][ntile] fp32 out

    __syncthreads();

    for (int e = 0; e < 4; ++e) {
        v8f c00 = {}, c01 = {}, c10 = {}, c11 = {};
#pragma unroll
        for (int kc = 0; kc < 9; ++kc) {
            const int ky = kc / 3, kx = kc % 3;
            // B fragments: K = ci (lanes0-15: 0..15, lanes16-31: 16..31)
            v16bf b0 = *(const v16bf*)&xlds[rsel + ky][cb + n + kx][lHi * 16];
            v16bf b1 = *(const v16bf*)&xlds[rsel + ky][cb + 16 + n + kx][lHi * 16];
            v16bf a0 = *(const v16bf*)&alds[e][0][kc][lane][0];
            v16bf a1 = *(const v16bf*)&alds[e][1][kc][lane][0];
            c00 = __builtin_amdgcn_wmma_f32_16x16x32_bf16(
                false, a0, false, b0, (short)0, c00, false, false);
            c01 = __builtin_amdgcn_wmma_f32_16x16x32_bf16(
                false, a0, false, b1, (short)0, c01, false, false);
            c10 = __builtin_amdgcn_wmma_f32_16x16x32_bf16(
                false, a1, false, b0, (short)0, c10, false, false);
            c11 = __builtin_amdgcn_wmma_f32_16x16x32_bf16(
                false, a1, false, b1, (short)0, c11, false, false);
        }

        // ---- fused BN + SiLU + routing-weighted accumulate ----
        // C/D layout: VGPR r, lanes0-15 -> M=r, lanes16-31 -> M=8+r
        const float we = wroute[e];
#pragma unroll
        for (int r = 0; r < 8; ++r) {
            int co0 = r + lHi * 8;
            int co1 = co0 + 16;
            float s0 = sclds[e][co0], t0 = shlds[e][co0];
            float s1 = sclds[e][co1], t1 = shlds[e][co1];
            float y;
            y = c00[r] * s0 + t0;  y = y / (1.0f + __expf(-y));  o00[r] += we * y;
            y = c01[r] * s0 + t0;  y = y / (1.0f + __expf(-y));  o01[r] += we * y;
            y = c10[r] * s1 + t1;  y = y / (1.0f + __expf(-y));  o10[r] += we * y;
            y = c11[r] * s1 + t1;  y = y / (1.0f + __expf(-y));  o11[r] += we * y;
        }
    }

    // ---- store: out[b][co][h][pix] fp32 ----
#pragma unroll
    for (int r = 0; r < 8; ++r) {
        int co0 = r + lHi * 8;
        size_t base0 = ((size_t)(b * 32 + co0) * Hh + h) * Ww;
        size_t base1 = ((size_t)(b * 32 + co0 + 16) * Hh + h) * Ww;
        out[base0 + cb + n]      = o00[r];
        out[base0 + cb + 16 + n] = o01[r];
        out[base1 + cb + n]      = o10[r];
        out[base1 + cb + 16 + n] = o11[r];
    }
}

extern "C" void kernel_launch(void* const* d_in, const int* in_sizes, int n_in,
                              void* d_out, int out_size, void* d_ws, size_t ws_size,
                              hipStream_t stream) {
    (void)in_sizes; (void)n_in; (void)out_size; (void)ws_size;
    const float* x   = (const float*)d_in[0];
    const float* rw  = (const float*)d_in[1];
    const int*   ri  = (const int*)  d_in[2];
    const float* Wc  = (const float*)d_in[3];
    const float* g   = (const float*)d_in[4];
    const float* be  = (const float*)d_in[5];
    const float* mn  = (const float*)d_in[6];
    const float* vr  = (const float*)d_in[7];
    float* out = (float*)d_out;
    unsigned short* wsw = (unsigned short*)d_ws;   // needs 73,728 B scratch

    // 1) swizzle weights to bf16 A-fragment layout in workspace (stateless,
    //    recomputed every call; same stream => ordered before main kernel)
    moe_prep_weights<<<dim3((WREC + BDIM - 1) / BDIM), dim3(BDIM), 0, stream>>>(Wc, wsw);

    // 2) main fused conv+BN+SiLU+MoE-combine kernel
    dim3 grid(Bn * (Hh / 2));   // one block per (sample, row-pair) = 4096
    moe_conv_bn_silu_kernel<<<grid, dim3(BDIM), 0, stream>>>(
        x, rw, ri, wsw, g, be, mn, vr, out);
}
